// ACE_48808008352089
// MI455X (gfx1250) — matrix operand order repaired and checked by reading, not compiled, not run on hardware
//
#include <hip/hip_runtime.h>

typedef unsigned short ushort_t;
typedef unsigned int uint_t;
typedef __attribute__((ext_vector_type(16))) __bf16 v16bf;
typedef __attribute__((ext_vector_type(8)))  float  v8f;

#define BB 4
#define CC 128
#define HH 256
#define WW 256
#define LL 19
#define SS 512
#define HWPLANE (HH*WW)
#define NRED (BB*HH*WW)

__device__ __forceinline__ float bf2f(ushort_t h) {
    union { uint_t u; float f; } v; v.u = ((uint_t)h) << 16; return v.f;
}
__device__ __forceinline__ ushort_t f2bf(float f) {
    union { float f; uint_t u; } v; v.f = f;
    uint_t r = v.u + 0x7FFFu + ((v.u >> 16) & 1u);
    return (ushort_t)(r >> 16);
}
__device__ __forceinline__ float sigm(float x) { return 1.0f / (1.0f + __expf(-x)); }

union FragB { uint4 q[2]; v16bf v; ushort_t e[16]; };

// ---------------- K0: transpose noise (B,W,H,1) -> noiseT[b][h][w] ----------
__global__ void k0_noiset(const float* __restrict__ noise, float* __restrict__ noiseT) {
    int idx = blockIdx.x * 256 + threadIdx.x;
    if (idx >= BB * HWPLANE) return;
    int b = idx >> 16, p = idx & 65535;
    int h = p >> 8, w = p & 255;
    noiseT[idx] = noise[b * HWPLANE + w * HH + h];
}

// ---------------- K1: mu[b][j][o] = relu(sum_d sc[b][j][d]*Wfc[j][o][d]+bfc) -
__global__ void k1_mu(const float* __restrict__ sc, const float* __restrict__ Wfc,
                      const float* __restrict__ bfc, float* __restrict__ mu) {
    __shared__ float scs[BB * SS];
    int j = blockIdx.x, tid = threadIdx.x;
    for (int i = tid; i < BB * SS; i += 256)
        scs[i] = sc[((i >> 9) * LL + j) * SS + (i & 511)];
    __syncthreads();
    for (int rep = 0; rep < 2; ++rep) {
        int o = tid + rep * 256;
        float acc0 = 0.f, acc1 = 0.f, acc2 = 0.f, acc3 = 0.f;
        const float4* wr = (const float4*)(Wfc + ((size_t)j * SS + o) * SS);
        for (int d4 = 0; d4 < SS / 4; ++d4) {
            float4 wv = wr[d4];
            int d = d4 * 4;
            acc0 += wv.x * scs[0 * SS + d] + wv.y * scs[0 * SS + d + 1] + wv.z * scs[0 * SS + d + 2] + wv.w * scs[0 * SS + d + 3];
            acc1 += wv.x * scs[1 * SS + d] + wv.y * scs[1 * SS + d + 1] + wv.z * scs[1 * SS + d + 2] + wv.w * scs[1 * SS + d + 3];
            acc2 += wv.x * scs[2 * SS + d] + wv.y * scs[2 * SS + d + 1] + wv.z * scs[2 * SS + d + 2] + wv.w * scs[2 * SS + d + 3];
            acc3 += wv.x * scs[3 * SS + d] + wv.y * scs[3 * SS + d + 1] + wv.z * scs[3 * SS + d + 2] + wv.w * scs[3 * SS + d + 3];
        }
        float bia = bfc[j * SS + o];
        mu[(0 * LL + j) * SS + o] = fmaxf(acc0 + bia, 0.f);
        mu[(1 * LL + j) * SS + o] = fmaxf(acc1 + bia, 0.f);
        mu[(2 * LL + j) * SS + o] = fmaxf(acc2 + bia, 0.f);
        mu[(3 * LL + j) * SS + o] = fmaxf(acc3 + bia, 0.f);
    }
}

// ---------------- K2: tap tables Tg/Tb[bj][t][o] = sum_c w[o][c][t]*mu[bj][c]
__global__ void k2_taps(const float* __restrict__ mu, const float* __restrict__ wcg,
                        const float* __restrict__ wcb, float* __restrict__ Tg,
                        float* __restrict__ Tb) {
    __shared__ float mus[SS];
    int bj = blockIdx.x, t = blockIdx.y, tid = threadIdx.x;
    for (int i = tid; i < SS; i += 128) mus[i] = mu[bj * SS + i];
    __syncthreads();
    int o = tid;
    float ag = 0.f, ab = 0.f;
    for (int c = 0; c < SS; ++c) {
        float m = mus[c];
        ag += wcg[(o * SS + c) * 9 + t] * m;
        ab += wcb[(o * SS + c) * 9 + t] * m;
    }
    Tg[(bj * 9 + t) * CC + o] = ag;
    Tb[(bj * 9 + t) * CC + o] = ab;
}

// ---------------- K3: partial per-channel sums of xn = x + noiseT*nv[c] -----
__global__ void k3_part(const float* __restrict__ x, const float* __restrict__ noiseT,
                        const float* __restrict__ nv, float* __restrict__ p1,
                        float* __restrict__ p2) {
    __shared__ float r1[256], r2[256];
    int c = blockIdx.x, slice = blockIdx.y, tid = threadIdx.x;
    float nvc = nv[c];
    float s1 = 0.f, s2 = 0.f;
    int F0 = slice * 4096;
    for (int r = 0; r < 4; ++r) {
        int F = F0 + (tid + r * 256) * 4;
        int b = F >> 16, p = F & 65535;
        float4 xv = *(const float4*)(x + ((size_t)(b * CC + c)) * HWPLANE + p);
        float4 nvv = *(const float4*)(noiseT + F);
        float a0 = xv.x + nvv.x * nvc, a1 = xv.y + nvv.y * nvc;
        float a2 = xv.z + nvv.z * nvc, a3 = xv.w + nvv.w * nvc;
        s1 += a0 + a1 + a2 + a3;
        s2 += a0 * a0 + a1 * a1 + a2 * a2 + a3 * a3;
    }
    r1[tid] = s1; r2[tid] = s2;
    __syncthreads();
    for (int off = 128; off > 0; off >>= 1) {
        if (tid < off) { r1[tid] += r1[tid + off]; r2[tid] += r2[tid + off]; }
        __syncthreads();
    }
    if (tid == 0) { p1[c * 64 + slice] = r1[0]; p2[c * 64 + slice] = r2[0]; }
}

// ---------------- K3b: finalize mean / inv-std -----------------------------
__global__ void k3b_stats(const float* __restrict__ p1, const float* __restrict__ p2,
                          float* __restrict__ stats) {
    int c = threadIdx.x;
    if (c >= CC) return;
    float s1 = 0.f, s2 = 0.f;
    for (int s = 0; s < 64; ++s) { s1 += p1[c * 64 + s]; s2 += p2[c * 64 + s]; }
    float mean = s1 / (float)NRED;
    float var = s2 / (float)NRED - mean * mean;
    stats[c] = mean;
    stats[CC + c] = rsqrtf(var + 1e-5f);
}

// ---------------- K4: actv bf16 [b][h][w][m] via LDS lookup table -----------
__global__ void k4_actv(const int* __restrict__ labels, const float* __restrict__ wsh,
                        const float* __restrict__ bsh_g, ushort_t* __restrict__ actv) {
    __shared__ ushort_t tab[CC * LL * 9];      // w_sh as bf16, layout [m][j][t]
    __shared__ int labs[3 * WW];
    __shared__ float bshs[CC];
    __shared__ __align__(16) ushort_t rowbuf[WW * 136];
    int h = blockIdx.x, b = blockIdx.y, tid = threadIdx.x;
    for (int i = tid; i < CC * LL * 9; i += 256) tab[i] = f2bf(wsh[i]);
    for (int i = tid; i < 3 * WW; i += 256) {
        int rr = i >> 8, ww = i & 255, hh = h + rr - 1;
        labs[i] = (hh >= 0 && hh < HH) ? labels[(b * HH + hh) * WW + ww] : -1;
    }
    if (tid < CC) bshs[tid] = bsh_g[tid];
    __syncthreads();
    int w = tid;
    int lt[9];
#pragma unroll
    for (int t = 0; t < 9; ++t) {
        int dx = t % 3 - 1, ww = w + dx;
        lt[t] = (ww >= 0 && ww < WW) ? labs[(t / 3) * WW + ww] : -1;
    }
    for (int m = 0; m < CC; ++m) {
        float acc = bshs[m];
        const ushort_t* row = &tab[m * (LL * 9)];
#pragma unroll
        for (int t = 0; t < 9; ++t) {
            int l = lt[t];
            int li = l < 0 ? 0 : l;
            float v = bf2f(row[li * 9 + t]);
            acc += (l >= 0) ? v : 0.f;
        }
        rowbuf[w * 136 + m] = f2bf(fmaxf(acc, 0.f));
    }
    __syncthreads();
    // coalesced write-out: 256*128 halves = 4096 x 16B
    for (int i = tid; i < WW * (CC / 8); i += 256) {
        int w2 = i >> 4, part = i & 15;
        uint4 v = *(const uint4*)&rowbuf[w2 * 136 + part * 8];
        *(uint4*)(actv + (((size_t)(b * HH + h) * WW + w2) * CC + part * 8)) = v;
    }
}

// ---------------- P1: pack spade weights into WMMA-A fragment order ---------
// layout: ((t*4+kc)*8+mt)*32+lane)*16+e ; K(e,laneHi) = e+laneHi*16
__global__ void p1_packw(const float* __restrict__ wg, const float* __restrict__ wb,
                         const float* __restrict__ gbl, const float* __restrict__ bbl,
                         ushort_t* __restrict__ Ag, ushort_t* __restrict__ Ab) {
    int idx = blockIdx.x * 256 + threadIdx.x;
    int e = idx & 15, lane = (idx >> 4) & 31, mt = (idx >> 9) & 7;
    int kc = (idx >> 12) & 3, t = idx >> 14;
    if (t >= 9) return;
    int o = mt * 16 + (lane & 15);
    int laneHi = (lane >> 4) & 1;
    int c = kc * 32 + e + laneHi * 16;
    float sg = 1.f - sigm(gbl[0]);
    float sb = 1.f - sigm(bbl[0]);
    Ag[idx] = f2bf(wg[(o * CC + c) * 9 + t] * sg);
    Ab[idx] = f2bf(wb[(o * CC + c) * 9 + t] * sb);
}

// ---------------- P2: pack Tg/Tb into WMMA-A fragments (K=32, j=0..18) ------
// layout: (((b*9+t)*8+mt)*32+lane)*16+e
__global__ void p2_packt(const float* __restrict__ Tg, const float* __restrict__ Tb,
                         const float* __restrict__ gbl, const float* __restrict__ bbl,
                         ushort_t* __restrict__ ATg, ushort_t* __restrict__ ATb) {
    int idx = blockIdx.x * 256 + threadIdx.x;
    int e = idx & 15, lane = (idx >> 4) & 31, mt = (idx >> 9) & 7;
    int bt = idx >> 12;
    int t = bt % 9, b = bt / 9;
    if (b >= BB) return;
    int o = mt * 16 + (lane & 15);
    int laneHi = (lane >> 4) & 1;
    int j = e + laneHi * 16;
    float ga = sigm(gbl[0]), ba = sigm(bbl[0]);
    float vg = 0.f, vb = 0.f;
    if (j < LL) {
        vg = Tg[((b * LL + j) * 9 + t) * CC + o] * ga;
        vb = Tb[((b * LL + j) * 9 + t) * CC + o] * ba;
    }
    ATg[idx] = f2bf(vg);
    ATb[idx] = f2bf(vb);
}

// ---------------- P3: blended biases ----------------------------------------
__global__ void p3_bias(const float* __restrict__ bcg, const float* __restrict__ bg,
                        const float* __restrict__ bcb, const float* __restrict__ bbv,
                        const float* __restrict__ gbl, const float* __restrict__ bbl,
                        float* __restrict__ bias) {
    int o = threadIdx.x;
    if (o >= CC) return;
    float ga = sigm(gbl[0]), ba = sigm(bbl[0]);
    bias[o] = ga * bcg[o] + (1.f - ga) * bg[o];
    bias[CC + o] = ba * bcb[o] + (1.f - ba) * bbv[o];
}

// ---------------- K5: fused WMMA conv + one-hot avg GEMM + normalize --------
__global__ __launch_bounds__(256)
void k5_main(const float* __restrict__ x, const int* __restrict__ labels,
             const float* __restrict__ noiseT, const float* __restrict__ nvar,
             const ushort_t* __restrict__ actv,
             const ushort_t* __restrict__ Ag, const ushort_t* __restrict__ Ab,
             const ushort_t* __restrict__ ATg, const ushort_t* __restrict__ ATb,
             const float* __restrict__ stats, const float* __restrict__ bias,
             float* __restrict__ out) {
    __shared__ __align__(16) ushort_t tile[180 * 136]; // halo actv: [10r][18c][128m], pitch 136
    __shared__ int labs[180];
    const int b = blockIdx.z;
    const int h0 = blockIdx.y * 8;
    const int w0 = blockIdx.x * 16;
    const int tid = threadIdx.x;
    // labels halo
    for (int i = tid; i < 180; i += 256) {
        int r = i / 18, cp = i % 18;
        int hh = h0 + r - 1, ww = w0 + cp - 1;
        labs[i] = (hh >= 0 && hh < HH && ww >= 0 && ww < WW) ? labels[(b * HH + hh) * WW + ww] : -1;
    }
    // actv halo tile: async global->LDS DMA (ASYNCcnt path), zeros for OOB halo
    for (int i = tid; i < 180 * 16; i += 256) {
        int px = i >> 4, part = i & 15;
        int r = px / 18, cp = px % 18;
        int hh = h0 + r - 1, ww = w0 + cp - 1;
        ushort_t* ldst = &tile[px * 136 + part * 8];
        if (hh >= 0 && hh < HH && ww >= 0 && ww < WW) {
            const ushort_t* gsrc = actv + (((size_t)(b * HH + hh) * WW + ww) * CC + part * 8);
            uint_t laddr = (uint_t)(size_t)ldst;
            asm volatile("global_load_async_to_lds_b128 %0, %1, off"
                         :: "v"(laddr), "v"(gsrc) : "memory");
        } else {
            *(uint4*)ldst = make_uint4(0, 0, 0, 0);
        }
    }
    // warm L2/L0 with the x-tile needed by the epilogue while WMMAs run
    for (int i = tid; i < CC * 8; i += 256) {
        int o = i >> 3, rr = i & 7;
        __builtin_prefetch(x + (((size_t)(b * CC + o) * HH) + h0 + rr) * WW + w0, 0, 3);
    }
    asm volatile("s_wait_asynccnt 0x0" ::: "memory");
    __syncthreads();

    const int wid = tid >> 5;           // wave id = M tile (o-base = wid*16)
    const int lane = tid & 31;
    const int laneN = lane & 15;
    const int laneHi = (lane >> 4) & 1;

    v8f cg[8], cb[8];
#pragma unroll
    for (int nt = 0; nt < 8; ++nt)
#pragma unroll
        for (int r = 0; r < 8; ++r) {
            int o = wid * 16 + r + laneHi * 8;
            cg[nt][r] = bias[o];
            cb[nt][r] = bias[CC + o];
        }

    // spade path: 9 taps x 4 K-chunks, shared B-frag for both convs
#pragma unroll 1
    for (int t = 0; t < 9; ++t) {
        int dy = t / 3 - 1, dx = t % 3 - 1;
#pragma unroll 1
        for (int kc = 0; kc < 4; ++kc) {
            v16bf ag = *(const v16bf*)(Ag + ((((t * 4 + kc) * 8 + wid) * 32 + lane) * 16));
            v16bf ab = *(const v16bf*)(Ab + ((((t * 4 + kc) * 8 + wid) * 32 + lane) * 16));
            int c0 = kc * 32 + laneHi * 16;
#pragma unroll
            for (int nt = 0; nt < 8; ++nt) {
                int pr = nt + 1 + dy, pc = laneN + 1 + dx;
                const uint4* p = (const uint4*)&tile[(pr * 18 + pc) * 136 + c0];
                FragB f; f.q[0] = p[0]; f.q[1] = p[1];
                cg[nt] = __builtin_amdgcn_wmma_f32_16x16x32_bf16(false, ag, false, f.v,
                                                                 (short)0, cg[nt], false, false);
                cb[nt] = __builtin_amdgcn_wmma_f32_16x16x32_bf16(false, ab, false, f.v,
                                                                 (short)0, cb[nt], false, false);
            }
        }
    }
    // avg path: one-hot labels GEMM, K=32 (j=0..18 valid)
#pragma unroll 1
    for (int t = 0; t < 9; ++t) {
        int dy = t / 3 - 1, dx = t % 3 - 1;
        v16bf ag = *(const v16bf*)(ATg + ((((b * 9 + t) * 8 + wid) * 32 + lane) * 16));
        v16bf ab = *(const v16bf*)(ATb + ((((b * 9 + t) * 8 + wid) * 32 + lane) * 16));
        int jb = laneHi * 16;
#pragma unroll
        for (int nt = 0; nt < 8; ++nt) {
            int pr = nt + 1 + dy, pc = laneN + 1 + dx;
            int lb = labs[pr * 18 + pc];
            FragB f;
#pragma unroll
            for (int k = 0; k < 16; ++k) f.e[k] = (lb == jb + k) ? (ushort_t)0x3F80 : (ushort_t)0;
            cg[nt] = __builtin_amdgcn_wmma_f32_16x16x32_bf16(false, ag, false, f.v,
                                                             (short)0, cg[nt], false, false);
            cb[nt] = __builtin_amdgcn_wmma_f32_16x16x32_bf16(false, ab, false, f.v,
                                                             (short)0, cb[nt], false, false);
        }
    }
    // epilogue: normalize + apply gamma/beta, write output
#pragma unroll 1
    for (int nt = 0; nt < 8; ++nt) {
        int hh = h0 + nt, ww = w0 + laneN;
        float nz = noiseT[(b * HH + hh) * WW + ww];
#pragma unroll
        for (int r = 0; r < 8; ++r) {
            int o = wid * 16 + r + laneHi * 8;
            float mean = stats[o], istd = stats[CC + o], nvc = nvar[o];
            size_t idx = (((size_t)(b * CC + o) * HH) + hh) * WW + ww;
            float xn = x[idx] + nz * nvc;
            float nrm = (xn - mean) * istd;
            out[idx] = nrm * (1.f + cg[nt][r]) + cb[nt][r];
        }
    }
}

extern "C" void kernel_launch(void* const* d_in, const int* in_sizes, int n_in,
                              void* d_out, int out_size, void* d_ws, size_t ws_size,
                              hipStream_t stream) {
    const float* x      = (const float*)d_in[0];
    const int*   labels = (const int*)d_in[1];
    const float* noise  = (const float*)d_in[2];
    const float* sc     = (const float*)d_in[3];
    const float* nvar   = (const float*)d_in[4];
    const float* w_sh   = (const float*)d_in[5];
    const float* b_sh   = (const float*)d_in[6];
    const float* w_g    = (const float*)d_in[7];
    const float* b_g    = (const float*)d_in[8];
    const float* w_b    = (const float*)d_in[9];
    const float* b_b    = (const float*)d_in[10];
    const float* Wfc    = (const float*)d_in[11];
    const float* bfc    = (const float*)d_in[12];
    const float* w_cg   = (const float*)d_in[13];
    const float* b_cg   = (const float*)d_in[14];
    const float* w_cb   = (const float*)d_in[15];
    const float* b_cb   = (const float*)d_in[16];
    const float* gbl    = (const float*)d_in[17];
    const float* bbl    = (const float*)d_in[18];
    float* out = (float*)d_out;

    char* ws = (char*)d_ws;
    float* noiseT = (float*)(ws + 0);                 // 1,048,576 B
    float* mu     = (float*)(ws + 1048576);           //   155,648 B
    float* Tg     = (float*)(ws + 1204224);           //   350,208 B
    float* Tb     = (float*)(ws + 1554432);           //   350,208 B
    float* part1  = (float*)(ws + 1904640);           //    32,768 B
    float* part2  = (float*)(ws + 1937408);           //    32,768 B
    float* stats  = (float*)(ws + 1970176);           //     1,024 B
    float* bias   = (float*)(ws + 1971200);           //     1,024 B
    ushort_t* Ag  = (ushort_t*)(ws + 1972224);        //   294,912 B
    ushort_t* Ab  = (ushort_t*)(ws + 2267136);        //   294,912 B
    ushort_t* ATg = (ushort_t*)(ws + 2562048);        //   294,912 B
    ushort_t* ATb = (ushort_t*)(ws + 2856960);        //   294,912 B
    ushort_t* actv= (ushort_t*)(ws + 3151872);        // 67,108,864 B

    k0_noiset<<<(BB * HWPLANE + 255) / 256, 256, 0, stream>>>(noise, noiseT);
    k1_mu<<<LL, 256, 0, stream>>>(sc, Wfc, bfc, mu);
    k2_taps<<<dim3(BB * LL, 9), 128, 0, stream>>>(mu, w_cg, w_cb, Tg, Tb);
    k3_part<<<dim3(CC, 64), 256, 0, stream>>>(x, noiseT, nvar, part1, part2);
    k3b_stats<<<1, 128, 0, stream>>>(part1, part2, stats);
    k4_actv<<<dim3(HH, BB), 256, 0, stream>>>(labels, w_sh, b_sh, actv);
    p1_packw<<<576, 256, 0, stream>>>(w_g, w_b, gbl, bbl, Ag, Ab);
    p2_packt<<<576, 256, 0, stream>>>(Tg, Tb, gbl, bbl, ATg, ATb);
    p3_bias<<<1, 128, 0, stream>>>(b_cg, b_g, b_cb, b_b, gbl, bbl, bias);
    k5_main<<<dim3(WW / 16, HH / 8, BB), 256, 0, stream>>>(
        x, labels, noiseT, nvar, actv, Ag, Ab, ATg, ATb, stats, bias, out);
}
